// WaveNet_45346264711374
// MI455X (gfx1250) — matrix operand order repaired and checked by reading, not compiled, not run on hardware
//
#include <hip/hip_runtime.h>

// ---------------------------------------------------------------------------
// WaveNet forward for MI455X (gfx1250, wave32, WMMA bf16 16x16x32 -> f32 acc)
// ---------------------------------------------------------------------------
#define BB 4
#define TT 8192
#define RR 64
#define SS 256
#define OO 256
#define CCH 80
#define LL 24
#define KPAD 224   // 64 (x[t-d]) + 64 (x[t]) + 80 (cond) + 16 zero-pad = 7*32

typedef __attribute__((ext_vector_type(16))) __bf16 bf16x16;
typedef __attribute__((ext_vector_type(8)))  __bf16 bf16x8;
typedef __attribute__((ext_vector_type(8)))  float  f32x8;

__device__ __forceinline__ f32x8 wmma_bf16(bf16x16 a, bf16x16 b, f32x8 c) {
  return __builtin_amdgcn_wmma_f32_16x16x32_bf16(false, a, false, b, (short)0, c,
                                                 false, false);
}

// A-matrix fragment (16x32 bf16) for wave32, per ISA table:
// lane L: row m = base + (L&15); halves 0..7 = k in [32s + 8*(L>>4), +8),
//                                halves 8..15 = k in [32s + 16 + 8*(L>>4), +8)
__device__ __forceinline__ bf16x16 load_afrag(const __bf16* __restrict__ wrow,
                                              int kbase, int lh) {
  union { bf16x16 v; bf16x8 h[2]; } u;
  const __bf16* p = wrow + kbase + 8 * lh;
  u.h[0] = *(const bf16x8*)(p);
  u.h[1] = *(const bf16x8*)(p + 16);
  return u.v;
}

// B-matrix fragment (32x16 bf16): lane L: col n = L&15,
// halves 0..7 = k in [32s + 8*(L>>4), +8), halves 8..15 = k in [32s + 16 + 8*(L>>4), +8)
__device__ __forceinline__ bf16x16 load_bfrag(const __bf16* __restrict__ rowp,
                                              int kbase, int lh) {
  union { bf16x16 v; bf16x8 h[2]; } u;
  const __bf16* p = rowp + kbase + 8 * lh;
  u.h[0] = *(const bf16x8*)(p);
  u.h[1] = *(const bf16x8*)(p + 16);
  return u.v;
}

// ---------------------------------------------------------------------------
// Pack weights to bf16, build fused dilate+cond matrix and fused bias
// ---------------------------------------------------------------------------
__global__ __launch_bounds__(256) void pack_kernel(
    const float* __restrict__ dw, const float* __restrict__ cw,
    const float* __restrict__ sw, const float* __restrict__ rw,
    const float* __restrict__ ow, const float* __restrict__ ew,
    const float* __restrict__ db, const float* __restrict__ cb,
    __bf16* __restrict__ wh, __bf16* __restrict__ wsk, __bf16* __restrict__ wr,
    __bf16* __restrict__ w1, __bf16* __restrict__ w2, float* __restrict__ bh) {
  const int NH = LL * 128 * KPAD;      // 688128
  const int NS = LL * SS * RR;         // 393216
  const int NR = (LL - 1) * RR * RR;   // 94208
  const int NO = OO * SS;              // 65536
  const int NB = LL * 128;             // 3072
  const int total = NH + NS + NR + 2 * NO + NB;
  for (int i = blockIdx.x * 256 + threadIdx.x; i < total; i += gridDim.x * 256) {
    int j = i;
    if (j < NH) {
      int l = j / (128 * KPAD), r = j % (128 * KPAD);
      int o = r / KPAD, k = r % KPAD;
      int oc = l * 128 + o;
      float v;
      if (k < 64)        v = dw[(oc * 64 + k) * 2 + 0];        // tap on x[t-d]
      else if (k < 128)  v = dw[(oc * 64 + (k - 64)) * 2 + 1]; // tap on x[t]
      else if (k < 208)  v = cw[oc * 80 + (k - 128)];          // cond proj
      else               v = 0.f;                              // K pad
      wh[j] = (__bf16)v;
      continue;
    }
    j -= NH;
    if (j < NS) { wsk[j] = (__bf16)sw[j]; continue; }
    j -= NS;
    if (j < NR) { wr[j] = (__bf16)rw[j]; continue; }
    j -= NR;
    if (j < NO) { w1[j] = (__bf16)ow[j]; continue; }
    j -= NO;
    if (j < NO) { w2[j] = (__bf16)ew[j]; continue; }
    j -= NO;
    bh[j] = db[j] + cb[j];  // fused dilate_b + cond_b  (indexed l*128+o)
  }
}

// ---------------------------------------------------------------------------
// ConvTranspose1d upsample (80ch, K=1024, stride=256), output as bf16 [b][c][t]
// ---------------------------------------------------------------------------
__global__ __launch_bounds__(256) void upsample_kernel(
    const float* __restrict__ feat, const float* __restrict__ w,
    const float* __restrict__ bias, __bf16* __restrict__ condb) {
  int idx = blockIdx.x * 256 + threadIdx.x;  // BB*CCH*TT
  if (idx >= BB * CCH * TT) return;
  int t = idx & (TT - 1);
  int r = idx / TT;
  int oc = r % CCH, b = r / CCH;
  float acc = bias[oc];
  int f1 = t >> 8;          if (f1 > 31) f1 = 31;
  int f0 = (t - 768) >> 8;  if (f0 < 0)  f0 = 0;   // ceil((t-1023)/256)
  for (int f = f0; f <= f1; ++f) {
    int k = t - (f << 8);  // in [0,1024)
    const float* wp = w + (size_t)oc * 1024 + k;         // w[ic][oc][k]
    const float* fp = feat + (size_t)b * CCH * 32 + f;
#pragma unroll 4
    for (int ic = 0; ic < CCH; ++ic)
      acc += fp[(size_t)ic * 32] * wp[(size_t)ic * CCH * 1024];
  }
  condb[idx] = (__bf16)acc;
}

// ---------------------------------------------------------------------------
// Token embedding -> residual stream x (f32 + bf16 mirror), layout [b][c][t]
// ---------------------------------------------------------------------------
__global__ __launch_bounds__(256) void embed_kernel(
    const int* __restrict__ tok, const float* __restrict__ emb,
    float* __restrict__ xf, __bf16* __restrict__ xb) {
  int idx = blockIdx.x * 256 + threadIdx.x;  // BB*RR*TT
  if (idx >= BB * RR * TT) return;
  int t = idx & (TT - 1);
  int r = idx / TT;
  int c = r % RR, b = r / RR;
  int tk = tok[b * TT + t];
  float v = emb[tk * RR + c];
  xf[idx] = v;
  xb[idx] = (__bf16)v;
}

// ---------------------------------------------------------------------------
// One WaveNet layer. Block = 8 waves, tile = 64 timesteps of one batch.
//   h[128,64]   = Whcat[128,224] @ [x(t-d);x(t);cond(t)]  (bf16 WMMA, f32 acc)
//   acts        = tanh(h[:64]) * sigmoid(h[64:])
//   skip       += Wskip[256,64] @ acts
//   x_next      = Wres[64,64] @ acts + res_b + x          (if has_res)
// ---------------------------------------------------------------------------
__global__ __launch_bounds__(256) void layer_kernel(
    const __bf16* __restrict__ Wh, const __bf16* __restrict__ Wsk,
    const __bf16* __restrict__ Wr, const float* __restrict__ bh,
    const float* __restrict__ sb, const float* __restrict__ rb,
    const float* __restrict__ xf_in, const __bf16* __restrict__ xb_in,
    float* __restrict__ xf_out, __bf16* __restrict__ xb_out,
    const __bf16* __restrict__ condb, float* __restrict__ skipbuf,
    int d, int is_first, int has_res) {
  __shared__ __attribute__((aligned(32))) __bf16 uin[64][KPAD];  // [n][k]
  __shared__ float hbuf[128][64];                                // [m][n]
  __shared__ __attribute__((aligned(32))) __bf16 abuf[64][64];   // [n][c]

  const int tid = threadIdx.x;
  const int b = blockIdx.y;
  const int t0 = blockIdx.x * 64;

  const __bf16* xb_b = xb_in + (size_t)b * RR * TT;
  const __bf16* cd_b = condb + (size_t)b * CCH * TT;

  // ---- stage B-operand tile into LDS ([time][k]) ----
  for (int i = tid; i < 64 * 64; i += 256) {           // x[t-d]
    int n = i & 63, c = i >> 6;
    int t = t0 + n - d;
    uin[n][c] = (t >= 0) ? xb_b[(size_t)c * TT + t] : (__bf16)0.f;
  }
  for (int i = tid; i < 64 * 64; i += 256) {           // x[t]
    int n = i & 63, c = i >> 6;
    uin[n][64 + c] = xb_b[(size_t)c * TT + t0 + n];
  }
  for (int i = tid; i < 64 * 80; i += 256) {           // cond[t]
    int n = i & 63, c = i >> 6;
    uin[n][128 + c] = cd_b[(size_t)c * TT + t0 + n];
  }
  for (int i = tid; i < 64 * 16; i += 256) {           // zero pad
    int n = i & 63, c = i >> 6;
    uin[n][208 + c] = (__bf16)0.f;
  }
  __syncthreads();

  const int wv = tid >> 5, ln = tid & 31;
  const int lh = ln >> 4, lm = ln & 15;

  // prefetch next-phase weights while doing the h GEMM
  __builtin_prefetch(Wsk + (size_t)(wv * 16 + lm) * 64, 0, 1);
  __builtin_prefetch(Wr + (size_t)((wv & 3) * 16 + lm) * 64, 0, 1);

  // ---- h GEMM: wave wv owns M-tile wv (rows 16wv..16wv+15), 4 N-tiles ----
  {
    f32x8 a0 = {}, a1 = {}, a2 = {}, a3 = {};
    const __bf16* wrow = Wh + (size_t)(wv * 16 + lm) * KPAD;
#pragma unroll
    for (int s = 0; s < 7; ++s) {
      bf16x16 af = load_afrag(wrow, 32 * s, lh);
      a0 = wmma_bf16(af, load_bfrag(&uin[ 0 + lm][0], 32 * s, lh), a0);
      a1 = wmma_bf16(af, load_bfrag(&uin[16 + lm][0], 32 * s, lh), a1);
      a2 = wmma_bf16(af, load_bfrag(&uin[32 + lm][0], 32 * s, lh), a2);
      a3 = wmma_bf16(af, load_bfrag(&uin[48 + lm][0], 32 * s, lh), a3);
    }
    int m0 = wv * 16;
#pragma unroll
    for (int v = 0; v < 8; ++v) {
      int m = m0 + v + 8 * lh;
      float bias = bh[m];
      hbuf[m][ 0 + lm] = a0[v] + bias;
      hbuf[m][16 + lm] = a1[v] + bias;
      hbuf[m][32 + lm] = a2[v] + bias;
      hbuf[m][48 + lm] = a3[v] + bias;
    }
  }
  __syncthreads();

  // ---- gate: acts = tanh(h[:64]) * sigmoid(h[64:]) -> bf16 [n][c] ----
  for (int i = tid; i < 64 * 64; i += 256) {
    int n = i & 63, c = i >> 6;
    float ht = hbuf[c][n], hs = hbuf[c + 64][n];
    float sg = 1.f / (1.f + __expf(-hs));
    abuf[n][c] = (__bf16)(tanhf(ht) * sg);
  }
  __syncthreads();

  // ---- skip GEMM (M=256): wave wv owns M-tiles {wv, wv+8} ----
  float* skp = skipbuf + (size_t)b * SS * TT;
#pragma unroll
  for (int half = 0; half < 2; ++half) {
    int m0 = (wv + half * 8) * 16;
    f32x8 s0 = {}, s1 = {}, s2 = {}, s3 = {};
    const __bf16* wrow = Wsk + (size_t)(m0 + lm) * 64;
#pragma unroll
    for (int s = 0; s < 2; ++s) {
      bf16x16 af = load_afrag(wrow, 32 * s, lh);
      s0 = wmma_bf16(af, load_bfrag(&abuf[ 0 + lm][0], 32 * s, lh), s0);
      s1 = wmma_bf16(af, load_bfrag(&abuf[16 + lm][0], 32 * s, lh), s1);
      s2 = wmma_bf16(af, load_bfrag(&abuf[32 + lm][0], 32 * s, lh), s2);
      s3 = wmma_bf16(af, load_bfrag(&abuf[48 + lm][0], 32 * s, lh), s3);
    }
#pragma unroll
    for (int v = 0; v < 8; ++v) {
      int m = m0 + v + 8 * lh;
      float bias = sb[m];
      float vals[4] = {s0[v], s1[v], s2[v], s3[v]};
#pragma unroll
      for (int q = 0; q < 4; ++q) {
        size_t idx = (size_t)m * TT + t0 + q * 16 + lm;
        float prev = is_first ? 0.f : skp[idx];
        skp[idx] = prev + vals[q] + bias;
      }
    }
  }

  // ---- residual GEMM (M=64): 8 waves cover 4 M-tiles x 2 N-halves ----
  if (has_res) {
    int m0 = (wv & 3) * 16;
    int ntb = (wv >> 2) * 2;
    f32x8 r0 = {}, r1 = {};
    const __bf16* wrow = Wr + (size_t)(m0 + lm) * 64;
#pragma unroll
    for (int s = 0; s < 2; ++s) {
      bf16x16 af = load_afrag(wrow, 32 * s, lh);
      r0 = wmma_bf16(af, load_bfrag(&abuf[(ntb + 0) * 16 + lm][0], 32 * s, lh), r0);
      r1 = wmma_bf16(af, load_bfrag(&abuf[(ntb + 1) * 16 + lm][0], 32 * s, lh), r1);
    }
    const float* xin = xf_in + (size_t)b * RR * TT;
    float* xo = xf_out + (size_t)b * RR * TT;
    __bf16* xbo = xb_out + (size_t)b * RR * TT;
#pragma unroll
    for (int v = 0; v < 8; ++v) {
      int m = m0 + v + 8 * lh;
      float bias = rb[m];
      float vv[2] = {r0[v], r1[v]};
#pragma unroll
      for (int q = 0; q < 2; ++q) {
        size_t idx = (size_t)m * TT + t0 + (ntb + q) * 16 + lm;
        float nv = vv[q] + bias + xin[idx];
        xo[idx] = nv;
        xbo[idx] = (__bf16)nv;
      }
    }
  }
}

// ---------------------------------------------------------------------------
// Output head: relu(skip) -> W1 -> relu -> W2 -> shift right by 1
// ---------------------------------------------------------------------------
__global__ __launch_bounds__(256) void out_kernel(
    const float* __restrict__ skipbuf, const __bf16* __restrict__ W1,
    const __bf16* __restrict__ W2, float* __restrict__ out) {
  __shared__ __attribute__((aligned(32))) __bf16 u1[64][264];  // relu(skip) [n][k]
  __shared__ __attribute__((aligned(32))) __bf16 u2[64][264];  // relu(y1)   [n][k]
  const int tid = threadIdx.x;
  const int b = blockIdx.y;
  const int t0 = blockIdx.x * 64;
  const float* skp = skipbuf + (size_t)b * SS * TT;

  for (int i = tid; i < 64 * 256; i += 256) {
    int n = i & 63, k = i >> 6;
    float v = skp[(size_t)k * TT + t0 + n];
    u1[n][k] = (__bf16)fmaxf(v, 0.f);
  }
  __syncthreads();

  const int wv = tid >> 5, ln = tid & 31;
  const int lh = ln >> 4, lm = ln & 15;

  // GEMM1 (256x256 @ 256x64) -> relu -> u2
#pragma unroll
  for (int half = 0; half < 2; ++half) {
    int m0 = (wv + half * 8) * 16;
    f32x8 a0 = {}, a1 = {}, a2 = {}, a3 = {};
    const __bf16* wrow = W1 + (size_t)(m0 + lm) * 256;
#pragma unroll
    for (int s = 0; s < 8; ++s) {
      bf16x16 af = load_afrag(wrow, 32 * s, lh);
      a0 = wmma_bf16(af, load_bfrag(&u1[ 0 + lm][0], 32 * s, lh), a0);
      a1 = wmma_bf16(af, load_bfrag(&u1[16 + lm][0], 32 * s, lh), a1);
      a2 = wmma_bf16(af, load_bfrag(&u1[32 + lm][0], 32 * s, lh), a2);
      a3 = wmma_bf16(af, load_bfrag(&u1[48 + lm][0], 32 * s, lh), a3);
    }
#pragma unroll
    for (int v = 0; v < 8; ++v) {
      int m = m0 + v + 8 * lh;
      u2[ 0 + lm][m] = (__bf16)fmaxf(a0[v], 0.f);
      u2[16 + lm][m] = (__bf16)fmaxf(a1[v], 0.f);
      u2[32 + lm][m] = (__bf16)fmaxf(a2[v], 0.f);
      u2[48 + lm][m] = (__bf16)fmaxf(a3[v], 0.f);
    }
  }
  __syncthreads();

  float* ob = out + (size_t)b * OO * TT;
  if (blockIdx.x == 0) ob[(size_t)tid * TT] = 0.f;  // zero first timestep

  // GEMM2 -> shifted store
#pragma unroll
  for (int half = 0; half < 2; ++half) {
    int m0 = (wv + half * 8) * 16;
    f32x8 a0 = {}, a1 = {}, a2 = {}, a3 = {};
    const __bf16* wrow = W2 + (size_t)(m0 + lm) * 256;
#pragma unroll
    for (int s = 0; s < 8; ++s) {
      bf16x16 af = load_afrag(wrow, 32 * s, lh);
      a0 = wmma_bf16(af, load_bfrag(&u2[ 0 + lm][0], 32 * s, lh), a0);
      a1 = wmma_bf16(af, load_bfrag(&u2[16 + lm][0], 32 * s, lh), a1);
      a2 = wmma_bf16(af, load_bfrag(&u2[32 + lm][0], 32 * s, lh), a2);
      a3 = wmma_bf16(af, load_bfrag(&u2[48 + lm][0], 32 * s, lh), a3);
    }
#pragma unroll
    for (int v = 0; v < 8; ++v) {
      int m = m0 + v + 8 * lh;
      float vals[4] = {a0[v], a1[v], a2[v], a3[v]};
#pragma unroll
      for (int q = 0; q < 4; ++q) {
        int t = t0 + q * 16 + lm;
        if (t + 1 < TT) ob[(size_t)m * TT + t + 1] = vals[q];
      }
    }
  }
}

// ---------------------------------------------------------------------------
extern "C" void kernel_launch(void* const* d_in, const int* in_sizes, int n_in,
                              void* d_out, int out_size, void* d_ws,
                              size_t ws_size, hipStream_t stream) {
  const float* features = (const float*)d_in[0];
  const int* tokens = (const int*)d_in[1];
  const float* upw = (const float*)d_in[2];
  const float* upb = (const float*)d_in[3];
  const float* cw = (const float*)d_in[4];
  const float* cb = (const float*)d_in[5];
  const float* emb = (const float*)d_in[6];
  const float* dw = (const float*)d_in[7];
  const float* db = (const float*)d_in[8];
  const float* rw = (const float*)d_in[9];
  const float* rb = (const float*)d_in[10];
  const float* sw = (const float*)d_in[11];
  const float* sb = (const float*)d_in[12];
  const float* ow = (const float*)d_in[13];
  const float* ew = (const float*)d_in[14];

  char* base = (char*)d_ws;
  size_t off = 0;
  auto take = [&](size_t bytes) -> char* {
    char* p = base + off;
    off += (bytes + 255) & ~(size_t)255;
    return p;
  };
  __bf16* WH  = (__bf16*)take((size_t)LL * 128 * KPAD * 2);
  __bf16* WSK = (__bf16*)take((size_t)LL * SS * RR * 2);
  __bf16* WR  = (__bf16*)take((size_t)(LL - 1) * RR * RR * 2);
  __bf16* W1  = (__bf16*)take((size_t)OO * SS * 2);
  __bf16* W2  = (__bf16*)take((size_t)OO * OO * 2);
  float*  BH  = (float*)take((size_t)LL * 128 * 4);
  __bf16* CONDB = (__bf16*)take((size_t)BB * CCH * TT * 2);
  float*  XF[2]; __bf16* XB[2];
  XF[0] = (float*)take((size_t)BB * RR * TT * 4);
  XF[1] = (float*)take((size_t)BB * RR * TT * 4);
  XB[0] = (__bf16*)take((size_t)BB * RR * TT * 2);
  XB[1] = (__bf16*)take((size_t)BB * RR * TT * 2);
  float* SKIP = (float*)take((size_t)BB * SS * TT * 4);

  // 1) pack/convert weights
  pack_kernel<<<5116, 256, 0, stream>>>(dw, cw, sw, rw, ow, ew, db, cb,
                                        WH, WSK, WR, W1, W2, BH);
  // 2) upsample conditioning
  upsample_kernel<<<(BB * CCH * TT) / 256, 256, 0, stream>>>(features, upw, upb,
                                                             CONDB);
  // 3) embedding -> x0
  embed_kernel<<<(BB * RR * TT) / 256, 256, 0, stream>>>(tokens, emb, XF[0],
                                                         XB[0]);
  // 4) 24 sequential gated residual layers
  dim3 grid(TT / 64, BB);
  for (int l = 0; l < LL; ++l) {
    int d = 1 << (l & 7);
    int cur = l & 1, nxt = (l + 1) & 1;
    int has_res = (l < LL - 1) ? 1 : 0;
    layer_kernel<<<grid, 256, 0, stream>>>(
        WH + (size_t)l * 128 * KPAD, WSK + (size_t)l * SS * RR,
        WR + (size_t)(has_res ? l : 0) * RR * RR, BH + (size_t)l * 128,
        sb + (size_t)l * SS, rb + (size_t)(has_res ? l : 0) * RR,
        XF[cur], XB[cur], XF[nxt], XB[nxt], CONDB, SKIP, d,
        (l == 0) ? 1 : 0, has_res);
  }
  // 5) output head + shift
  out_kernel<<<grid, 256, 0, stream>>>(SKIP, W1, W2, (float*)d_out);
}
